// GatedMultiHeadAttention_90787018703086
// MI455X (gfx1250) — compile-verified, matrix operand order checked by reading, not run on hardware
//
#include <hip/hip_runtime.h>

// ---------------------------------------------------------------------------
// Gated multi-head attention for MI455X (gfx1250), bf16 WMMA + fp32 accum,
// async global->LDS double-buffered flash attention.
// B=8, N=1024, C=768, H=12, HD=64
// ---------------------------------------------------------------------------

typedef __attribute__((ext_vector_type(16))) __bf16 v16bf;
typedef __attribute__((ext_vector_type(8)))  float  v8f;

#define B_DIM   8
#define N_SEQ   1024
#define C_DIM   768
#define H_NUM   12
#define HD_DIM  64
#define SCALE   0.125f   // 1/sqrt(64)

// ---------------- bf16 helpers --------------------------------------------
static __device__ __forceinline__ __bf16 u16_as_bf16(unsigned short u) {
  union { unsigned short u; __bf16 h; } x; x.u = u; return x.h;
}
// native gfx1250 f32->bf16 convert (single VALU op)
static __device__ __forceinline__ unsigned short f32_to_bf16(float f) {
  union { __bf16 h; unsigned short u; } x; x.h = (__bf16)f; return x.u;
}

// K-packing of 16-bit A/B fragments (ISA 7.12.2): VGPR q holds K = k0,k0+1
static __device__ __forceinline__ int kpack(int q, int half) {
  return (q < 4 ? 2 * q : 16 + 2 * (q - 4)) + half * 8;
}

// A-style fragment: tile stored row-major [row][K], ld elements per row.
// Also used for B fragments when the B tile is stored transposed [n][k].
static __device__ __forceinline__ v16bf frag_rowmajor(
    const unsigned short* tile, int ld, int row, int half, int koff) {
  v16bf a;
  const unsigned short* p = tile + row * ld + koff;
#pragma unroll
  for (int q = 0; q < 8; ++q) {
    int k0 = kpack(q, half);
    a[2 * q]     = u16_as_bf16(p[k0]);
    a[2 * q + 1] = u16_as_bf16(p[k0 + 1]);
  }
  return a;
}

// B fragment from a tile stored [K][N] row-major (used for V in attention).
static __device__ __forceinline__ v16bf frag_kmajor(
    const unsigned short* tile, int ld, int col, int half, int koff) {
  v16bf a;
  const unsigned short* p = tile + koff * ld + col;
#pragma unroll
  for (int q = 0; q < 8; ++q) {
    int k0 = kpack(q, half);
    a[2 * q]     = u16_as_bf16(p[k0 * ld]);
    a[2 * q + 1] = u16_as_bf16(p[(k0 + 1) * ld]);
  }
  return a;
}

static __device__ __forceinline__ v8f wmma_bf16(v16bf a, v16bf b, v8f c) {
  return __builtin_amdgcn_wmma_f32_16x16x32_bf16(
      /*neg_a=*/false, a, /*neg_b=*/false, b,
      /*c_mod=*/(short)0, c, /*reuse_a=*/false, /*reuse_b=*/false);
}
static __device__ __forceinline__ v8f zero8() {
  v8f z = {0.f, 0.f, 0.f, 0.f, 0.f, 0.f, 0.f, 0.f};
  return z;
}

// ---------------- CDNA5 async global->LDS (ASYNCcnt-tracked) --------------
static __device__ __forceinline__ void async_ld_b128(void* lds, const void* g) {
  unsigned l = (unsigned)(size_t)lds;     // low 32 bits of LDS aperture addr
  asm volatile("global_load_async_to_lds_b128 %0, %1, off"
               :: "v"(l), "v"(g) : "memory");
}
#define WAIT_ASYNC(n) asm volatile("s_wait_asynccnt " #n ::: "memory")

// ---------------- LDS staging helpers (256-thread blocks) ------------------
#define LDSK 40  // padded K stride (u16) to spread banks

// 128x32 f32 tile (row-major, lda) -> bf16 LDS
static __device__ __forceinline__ void stage_a_f32(
    unsigned short* ldsA, const float* A, int lda, int tid) {
#pragma unroll
  for (int i = 0; i < 4; ++i) {
    int slot = i * 256 + tid;          // 1024 float4 slots
    int row  = slot >> 3;              // 8 float4 per 32-float row
    int c    = (slot & 7) << 2;
    const float4 f = *(const float4*)(A + (size_t)row * lda + c);
    unsigned short* d = ldsA + row * LDSK + c;
    d[0] = f32_to_bf16(f.x); d[1] = f32_to_bf16(f.y);
    d[2] = f32_to_bf16(f.z); d[3] = f32_to_bf16(f.w);
  }
}

// 32x128 f32 tile of W (row-major, ldb) -> transposed bf16 LDS [n][k]
static __device__ __forceinline__ void stage_bt_f32(
    unsigned short* ldsBt, const float* W, int ldb, int tid) {
#pragma unroll
  for (int i = 0; i < 4; ++i) {
    int slot = i * 256 + tid;          // 1024 float4 slots
    int krow = slot >> 5;              // 32 float4 per 128-float row
    int c    = (slot & 31) << 2;
    const float4 f = *(const float4*)(W + (size_t)krow * ldb + c);
    ldsBt[(c + 0) * LDSK + krow] = f32_to_bf16(f.x);
    ldsBt[(c + 1) * LDSK + krow] = f32_to_bf16(f.y);
    ldsBt[(c + 2) * LDSK + krow] = f32_to_bf16(f.z);
    ldsBt[(c + 3) * LDSK + krow] = f32_to_bf16(f.w);
  }
}

// prefetch next 32x128 weight tile into GL2 (global_prefetch_b8)
static __device__ __forceinline__ void prefetch_bt_f32(
    const float* W, int ldb, int tid) {
#pragma unroll
  for (int i = 0; i < 4; ++i) {
    int slot = i * 256 + tid;
    int krow = slot >> 5;
    int c    = (slot & 31) << 2;
    __builtin_prefetch(W + (size_t)krow * ldb + c, 0, 1);
  }
}

// 128x32 bf16 tile (row-major, lda) -> bf16 LDS (straight copy)
static __device__ __forceinline__ void stage_a_bf16(
    unsigned short* ldsA, const unsigned short* A, int lda, int tid) {
#pragma unroll
  for (int i = 0; i < 4; ++i) {
    int slot = i * 256 + tid;          // 1024 chunks of 4 u16
    int row  = slot >> 3;
    int c    = (slot & 7) << 2;
    uint2 d = *(const uint2*)(A + (size_t)row * lda + c);
    *(uint2*)(ldsA + row * LDSK + c) = d;
  }
}

// ===========================================================================
// Kernel 1: qkv = x @ qkv_w + qkv_b, scattered to Q/K/V [B,H,N,64] bf16
// grid = (2304/128, 8192/128), block = 256
// ===========================================================================
__global__ __launch_bounds__(256) void qkv_kernel(
    const float* __restrict__ x, const float* __restrict__ w,
    const float* __restrict__ bias,
    unsigned short* __restrict__ qo, unsigned short* __restrict__ ko,
    unsigned short* __restrict__ vo) {
  __shared__ unsigned short ldsA[128 * LDSK];
  __shared__ unsigned short ldsB[128 * LDSK];

  const int tid = threadIdx.x, lane = tid & 31, wid = tid >> 5;
  const int half = lane >> 4, lrow = lane & 15;
  const int wm = wid >> 1, wn = wid & 1;           // 4x2 wave grid
  const int blockM = blockIdx.y * 128;
  const int blockN = blockIdx.x * 128;

  v8f acc[2][4];
#pragma unroll
  for (int i = 0; i < 2; ++i)
#pragma unroll
    for (int j = 0; j < 4; ++j) acc[i][j] = zero8();

  for (int k0 = 0; k0 < C_DIM; k0 += 32) {
    stage_a_f32(ldsA, x + (size_t)blockM * C_DIM + k0, C_DIM, tid);
    stage_bt_f32(ldsB, w + (size_t)k0 * (3 * C_DIM) + blockN, 3 * C_DIM, tid);
    if (k0 + 32 < C_DIM)
      prefetch_bt_f32(w + (size_t)(k0 + 32) * (3 * C_DIM) + blockN, 3 * C_DIM, tid);
    __syncthreads();
    v16bf af[2], bfr[4];
#pragma unroll
    for (int i = 0; i < 2; ++i)
      af[i] = frag_rowmajor(ldsA + (wm * 32 + i * 16) * LDSK, LDSK, lrow, half, 0);
#pragma unroll
    for (int j = 0; j < 4; ++j)
      bfr[j] = frag_rowmajor(ldsB + (wn * 64 + j * 16) * LDSK, LDSK, lrow, half, 0);
#pragma unroll
    for (int i = 0; i < 2; ++i)
#pragma unroll
      for (int j = 0; j < 4; ++j) acc[i][j] = wmma_bf16(af[i], bfr[j], acc[i][j]);
    __syncthreads();
  }

  // Epilogue: D layout: n = lrow, m = r + 8*half  (per 16x16 frag)
#pragma unroll
  for (int i = 0; i < 2; ++i) {
#pragma unroll
    for (int j = 0; j < 4; ++j) {
      int gcol  = blockN + wn * 64 + j * 16 + lrow;   // [0,2304)
      int which = (gcol >= 2 * C_DIM) ? 2 : (gcol >= C_DIM ? 1 : 0);
      int rem   = gcol - which * C_DIM;
      int h     = rem >> 6, d = rem & 63;
      float bv  = bias[gcol];
      unsigned short* dst = (which == 0) ? qo : (which == 1) ? ko : vo;
#pragma unroll
      for (int r = 0; r < 8; ++r) {
        int m  = blockM + wm * 32 + i * 16 + r + half * 8;
        int b  = m >> 10, nq = m & 1023;
        size_t off = ((size_t)((b * H_NUM + h) * N_SEQ + nq)) * HD_DIM + d;
        dst[off] = f32_to_bf16(acc[i][j][r] + bv);
      }
    }
  }
}

// ===========================================================================
// Kernel 2: flash attention per (b,h), 64-query block per workgroup.
// grid = (N/64, B*H) = (16, 96), block = 128 (4 waves, 16 query rows each)
// K/V tiles double-buffered via global_load_async_to_lds_b128 (ASYNCcnt).
// ===========================================================================
__global__ __launch_bounds__(128) void attn_kernel(
    const unsigned short* __restrict__ Q, const unsigned short* __restrict__ K,
    const unsigned short* __restrict__ V, unsigned short* __restrict__ O) {
  __shared__ unsigned short Qs[64 * 64];
  __shared__ unsigned short Ks[2][64 * 64];
  __shared__ unsigned short Vs[2][64 * 64];
  __shared__ unsigned short Ps[4 * 16 * 64];   // per-wave P scratch

  const int qblk = blockIdx.x;        // 0..15
  const int bh   = blockIdx.y;        // 0..95
  const int b = bh / H_NUM, h = bh - b * H_NUM;

  const unsigned short* Qg = Q + (size_t)bh * N_SEQ * HD_DIM + (size_t)qblk * 64 * HD_DIM;
  const unsigned short* Kg = K + (size_t)bh * N_SEQ * HD_DIM;
  const unsigned short* Vg = V + (size_t)bh * N_SEQ * HD_DIM;

  const int tid = threadIdx.x, lane = tid & 31, wave = tid >> 5;
  const int half = lane >> 4, lrow = lane & 15;

  // Issue async K/V tile 0 (8 async ops per thread per tile)
  {
#pragma unroll
    for (int i = 0; i < 4; ++i) {
      int idx = i * 128 + tid;
      async_ld_b128(&Ks[0][idx * 8], Kg + idx * 8);
      async_ld_b128(&Vs[0][idx * 8], Vg + idx * 8);
    }
  }

  // Stage Q block (regular vector path; 64x64 bf16 = 512 uint4)
#pragma unroll
  for (int i = 0; i < 4; ++i) {
    int idx = i * 128 + tid;
    ((uint4*)Qs)[idx] = ((const uint4*)Qg)[idx];
  }
  __syncthreads();

  v16bf qf0 = frag_rowmajor(Qs + wave * 16 * 64, 64, lrow, half, 0);
  v16bf qf1 = frag_rowmajor(Qs + wave * 16 * 64, 64, lrow, half, 32);

  float mstate[8], lstate[8];
  v8f acc[4];
#pragma unroll
  for (int r = 0; r < 8; ++r) { mstate[r] = -3.0e38f; lstate[r] = 0.f; }
#pragma unroll
  for (int db = 0; db < 4; ++db) acc[db] = zero8();

  for (int kb = 0; kb < N_SEQ / 64; ++kb) {
    const int cur = kb & 1;
    if (kb + 1 < N_SEQ / 64) {
      // issue next tile into the other buffer, then wait for current tile:
      // the 8 newest async ops (per wave-lane) may remain outstanding.
      const unsigned short* kg = Kg + (size_t)(kb + 1) * 64 * HD_DIM;
      const unsigned short* vg = Vg + (size_t)(kb + 1) * 64 * HD_DIM;
#pragma unroll
      for (int i = 0; i < 4; ++i) {
        int idx = i * 128 + tid;
        async_ld_b128(&Ks[cur ^ 1][idx * 8], kg + idx * 8);
        async_ld_b128(&Vs[cur ^ 1][idx * 8], vg + idx * 8);
      }
      WAIT_ASYNC(0x8);
    } else {
      WAIT_ASYNC(0x0);
    }
    __syncthreads();   // current tile visible to all waves

    // S = (Q Kt) * SCALE : 16x64 per wave, four 16x16 tiles
    float s[4][8];
#pragma unroll
    for (int nb = 0; nb < 4; ++nb) {
      v16bf kf0 = frag_rowmajor(Ks[cur] + nb * 16 * 64, 64, lrow, half, 0);
      v16bf kf1 = frag_rowmajor(Ks[cur] + nb * 16 * 64, 64, lrow, half, 32);
      v8f c = zero8();
      c = wmma_bf16(qf0, kf0, c);
      c = wmma_bf16(qf1, kf1, c);
#pragma unroll
      for (int r = 0; r < 8; ++r) s[nb][r] = c[r] * SCALE;
    }

    // Online softmax per row (row m = r + 8*half; 16 n-lanes per half)
#pragma unroll
    for (int r = 0; r < 8; ++r) {
      float mx = fmaxf(fmaxf(s[0][r], s[1][r]), fmaxf(s[2][r], s[3][r]));
#pragma unroll
      for (int off = 1; off < 16; off <<= 1) mx = fmaxf(mx, __shfl_xor(mx, off, 32));
      float mnew = fmaxf(mstate[r], mx);
      float corr = __expf(mstate[r] - mnew);
      mstate[r] = mnew;
      float rs = 0.f;
#pragma unroll
      for (int nb = 0; nb < 4; ++nb) {
        float p = __expf(s[nb][r] - mnew);
        s[nb][r] = p;
        rs += p;
      }
#pragma unroll
      for (int off = 1; off < 16; off <<= 1) rs += __shfl_xor(rs, off, 32);
      lstate[r] = lstate[r] * corr + rs;
#pragma unroll
      for (int db = 0; db < 4; ++db) acc[db][r] *= corr;
    }

    // Re-shape P (C layout) -> A layout through per-wave LDS scratch
    unsigned short* pw = Ps + wave * 16 * 64;
#pragma unroll
    for (int nb = 0; nb < 4; ++nb)
#pragma unroll
      for (int r = 0; r < 8; ++r)
        pw[(r + half * 8) * 64 + nb * 16 + lrow] = f32_to_bf16(s[nb][r]);
    asm volatile("s_wait_dscnt 0x0" ::: "memory");

    v16bf pf0 = frag_rowmajor(pw, 64, lrow, half, 0);
    v16bf pf1 = frag_rowmajor(pw, 64, lrow, half, 32);

    // O += P V : V tile stored [key][d] -> k-major B fragments
#pragma unroll
    for (int db = 0; db < 4; ++db) {
      v16bf vf0 = frag_kmajor(Vs[cur], 64, db * 16 + lrow, half, 0);
      v16bf vf1 = frag_kmajor(Vs[cur], 64, db * 16 + lrow, half, 32);
      acc[db] = wmma_bf16(pf0, vf0, acc[db]);
      acc[db] = wmma_bf16(pf1, vf1, acc[db]);
    }
    __syncthreads();   // all waves done with this buffer before reissue
  }

  // Normalize and store to attn-out [B,N,C] bf16
#pragma unroll
  for (int db = 0; db < 4; ++db) {
    int d = db * 16 + lrow;
#pragma unroll
    for (int r = 0; r < 8; ++r) {
      int m  = r + half * 8;
      int nq = qblk * 64 + wave * 16 + m;
      float o = acc[db][r] / lstate[r];
      O[((size_t)(b * N_SEQ + nq)) * C_DIM + h * HD_DIM + d] = f32_to_bf16(o);
    }
  }
}

// ===========================================================================
// Kernel 3: proj = attn @ proj_w ; writes fp32 (for final multiply) + bf16
// grid = (768/128, 8192/128), block = 256
// ===========================================================================
__global__ __launch_bounds__(256) void proj_kernel(
    const unsigned short* __restrict__ A, const float* __restrict__ W,
    float* __restrict__ outf, unsigned short* __restrict__ outb) {
  __shared__ unsigned short ldsA[128 * LDSK];
  __shared__ unsigned short ldsB[128 * LDSK];

  const int tid = threadIdx.x, lane = tid & 31, wid = tid >> 5;
  const int half = lane >> 4, lrow = lane & 15;
  const int wm = wid >> 1, wn = wid & 1;
  const int blockM = blockIdx.y * 128;
  const int blockN = blockIdx.x * 128;

  v8f acc[2][4];
#pragma unroll
  for (int i = 0; i < 2; ++i)
#pragma unroll
    for (int j = 0; j < 4; ++j) acc[i][j] = zero8();

  for (int k0 = 0; k0 < C_DIM; k0 += 32) {
    stage_a_bf16(ldsA, A + (size_t)blockM * C_DIM + k0, C_DIM, tid);
    stage_bt_f32(ldsB, W + (size_t)k0 * C_DIM + blockN, C_DIM, tid);
    if (k0 + 32 < C_DIM)
      prefetch_bt_f32(W + (size_t)(k0 + 32) * C_DIM + blockN, C_DIM, tid);
    __syncthreads();
    v16bf af[2], bfr[4];
#pragma unroll
    for (int i = 0; i < 2; ++i)
      af[i] = frag_rowmajor(ldsA + (wm * 32 + i * 16) * LDSK, LDSK, lrow, half, 0);
#pragma unroll
    for (int j = 0; j < 4; ++j)
      bfr[j] = frag_rowmajor(ldsB + (wn * 64 + j * 16) * LDSK, LDSK, lrow, half, 0);
#pragma unroll
    for (int i = 0; i < 2; ++i)
#pragma unroll
      for (int j = 0; j < 4; ++j) acc[i][j] = wmma_bf16(af[i], bfr[j], acc[i][j]);
    __syncthreads();
  }

#pragma unroll
  for (int i = 0; i < 2; ++i) {
#pragma unroll
    for (int j = 0; j < 4; ++j) {
      int gcol = blockN + wn * 64 + j * 16 + lrow;
#pragma unroll
      for (int r = 0; r < 8; ++r) {
        int m = blockM + wm * 32 + i * 16 + r + half * 8;
        size_t off = (size_t)m * C_DIM + gcol;
        float v = acc[i][j][r];
        outf[off] = v;
        outb[off] = f32_to_bf16(v);
      }
    }
  }
}

// ===========================================================================
// Kernel 4: out = proj_f32 * sigmoid(proj_bf16 @ gate_w)
// grid = (768/128, 8192/128), block = 256
// ===========================================================================
__global__ __launch_bounds__(256) void gate_kernel(
    const unsigned short* __restrict__ A, const float* __restrict__ W,
    const float* __restrict__ projf, float* __restrict__ out) {
  __shared__ unsigned short ldsA[128 * LDSK];
  __shared__ unsigned short ldsB[128 * LDSK];

  const int tid = threadIdx.x, lane = tid & 31, wid = tid >> 5;
  const int half = lane >> 4, lrow = lane & 15;
  const int wm = wid >> 1, wn = wid & 1;
  const int blockM = blockIdx.y * 128;
  const int blockN = blockIdx.x * 128;

  v8f acc[2][4];
#pragma unroll
  for (int i = 0; i < 2; ++i)
#pragma unroll
    for (int j = 0; j < 4; ++j) acc[i][j] = zero8();

  for (int k0 = 0; k0 < C_DIM; k0 += 32) {
    stage_a_bf16(ldsA, A + (size_t)blockM * C_DIM + k0, C_DIM, tid);
    stage_bt_f32(ldsB, W + (size_t)k0 * C_DIM + blockN, C_DIM, tid);
    if (k0 + 32 < C_DIM)
      prefetch_bt_f32(W + (size_t)(k0 + 32) * C_DIM + blockN, C_DIM, tid);
    __syncthreads();
    v16bf af[2], bfr[4];
#pragma unroll
    for (int i = 0; i < 2; ++i)
      af[i] = frag_rowmajor(ldsA + (wm * 32 + i * 16) * LDSK, LDSK, lrow, half, 0);
#pragma unroll
    for (int j = 0; j < 4; ++j)
      bfr[j] = frag_rowmajor(ldsB + (wn * 64 + j * 16) * LDSK, LDSK, lrow, half, 0);
#pragma unroll
    for (int i = 0; i < 2; ++i)
#pragma unroll
      for (int j = 0; j < 4; ++j) acc[i][j] = wmma_bf16(af[i], bfr[j], acc[i][j]);
    __syncthreads();
  }

#pragma unroll
  for (int i = 0; i < 2; ++i) {
#pragma unroll
    for (int j = 0; j < 4; ++j) {
      int gcol = blockN + wn * 64 + j * 16 + lrow;
#pragma unroll
      for (int r = 0; r < 8; ++r) {
        int m = blockM + wm * 32 + i * 16 + r + half * 8;
        size_t off = (size_t)m * C_DIM + gcol;
        float g = 1.0f / (1.0f + __expf(-acc[i][j][r]));
        out[off] = projf[off] * g;
      }
    }
  }
}

// ===========================================================================
// Host launch
// ===========================================================================
extern "C" void kernel_launch(void* const* d_in, const int* in_sizes, int n_in,
                              void* d_out, int out_size, void* d_ws, size_t ws_size,
                              hipStream_t stream) {
  (void)in_sizes; (void)n_in; (void)out_size; (void)ws_size;
  const float* x      = (const float*)d_in[0];
  const float* qkv_w  = (const float*)d_in[1];
  const float* qkv_b  = (const float*)d_in[2];
  const float* gate_w = (const float*)d_in[3];
  const float* proj_w = (const float*)d_in[4];
  float* out = (float*)d_out;

  // Workspace layout: 5 bf16 arrays of 6,291,456 elems + 1 fp32 array (~88MB)
  const size_t NE = (size_t)B_DIM * N_SEQ * C_DIM;   // 6,291,456
  unsigned short* qs    = (unsigned short*)d_ws;
  unsigned short* ks    = qs + NE;
  unsigned short* vs    = ks + NE;
  unsigned short* attn  = vs + NE;
  unsigned short* projb = attn + NE;
  float*          projf = (float*)(projb + NE);

  // 1) QKV GEMM: [8192,768] x [768,2304]
  qkv_kernel<<<dim3((3 * C_DIM) / 128, (B_DIM * N_SEQ) / 128), 256, 0, stream>>>(
      x, qkv_w, qkv_b, qs, ks, vs);

  // 2) Flash attention: (b,h) x query-blocks
  attn_kernel<<<dim3(N_SEQ / 64, B_DIM * H_NUM), 128, 0, stream>>>(
      qs, ks, vs, attn);

  // 3) proj GEMM: [8192,768] x [768,768] -> fp32 + bf16
  proj_kernel<<<dim3(C_DIM / 128, (B_DIM * N_SEQ) / 128), 256, 0, stream>>>(
      attn, proj_w, projf, projb);

  // 4) gate GEMM + fused sigmoid-multiply epilogue
  gate_kernel<<<dim3(C_DIM / 128, (B_DIM * N_SEQ) / 128), 256, 0, stream>>>(
      projb, gate_w, projf, out);
}
